// MyDenseGAT_17626545782906
// MI455X (gfx1250) — matrix operand order repaired and checked by reading, not compiled, not run on hardware
//
#include <hip/hip_runtime.h>

// ---------------- problem constants ----------------
#define Bdim  4
#define Ndim  2048
#define Fdim  256
#define Hdim  4
#define Cdim  64
#define HCdim 256

typedef __attribute__((ext_vector_type(16))) __bf16 v16bf;
typedef __attribute__((ext_vector_type(8)))  float  v8f;

union Frag {
  v16bf v;
  uint4 q[2];
  unsigned short s[16];
};

__device__ __forceinline__ unsigned short f2bf(float x) {
  unsigned int u = __float_as_uint(x);
  unsigned int r = u + 0x7fffu + ((u >> 16) & 1u);   // RNE
  return (unsigned short)(r >> 16);
}

__device__ __forceinline__ v8f wmma_bf16(const Frag& a, const Frag& b, v8f c) {
  // D(16x16 f32) = A(16x32 bf16) x B(32x16 bf16) + C
  return __builtin_amdgcn_wmma_f32_16x16x32_bf16(false, a.v, false, b.v,
                                                 (short)0, c, false, false);
}

// LDS offset of a __shared__ object (generic ptr low 32 bits == LDS offset)
__device__ __forceinline__ unsigned ldsoff(const void* p) {
  return (unsigned)(unsigned long long)(uintptr_t)p;
}

// gfx1250 async global->LDS copies (ASYNCcnt path, no VGPR round-trip).
template <int OFF>
__device__ __forceinline__ void async_b128(unsigned ldsaddr, unsigned goff,
                                           unsigned long long base) {
  asm volatile("global_load_async_to_lds_b128 %0, %1, %2 offset:%c3"
               :: "v"(ldsaddr), "v"(goff), "s"(base), "i"(OFF) : "memory");
}
__device__ __forceinline__ void async_b32(unsigned ldsaddr, unsigned goff,
                                          unsigned long long base) {
  asm volatile("global_load_async_to_lds_b32 %0, %1, %2"
               :: "v"(ldsaddr), "v"(goff), "s"(base) : "memory");
}
__device__ __forceinline__ void wait_async0() {
  asm volatile("s_wait_asynccnt 0" ::: "memory");
}

// ---------------------------------------------------------------------------
// Kernel 1: f32 -> bf16; weights transposed (Wt[col][f]) for B-fragments.
// ---------------------------------------------------------------------------
__global__ void __launch_bounds__(256) k_convert(
    const float* __restrict__ x,
    const float* __restrict__ w0, const float* __restrict__ w1,
    const float* __restrict__ w2, const float* __restrict__ w3,
    const float* __restrict__ w4, const float* __restrict__ w5,
    unsigned short* __restrict__ xbf, unsigned short* __restrict__ wt)
{
  long long i = (long long)blockIdx.x * blockDim.x + threadIdx.x;
  const long long NX = (long long)Bdim * Ndim * Fdim;
  if (i < NX) { xbf[i] = f2bf(x[i]); return; }
  long long j = i - NX;
  if (j >= 6LL * Fdim * HCdim) return;
  int w   = (int)(j >> 16);
  int e   = (int)(j & 65535);
  int col = e >> 8, f = e & 255;
  const float* src = (w == 0) ? w0 : (w == 1) ? w1 : (w == 2) ? w2
                   : (w == 3) ? w3 : (w == 4) ? w4 : w5;
  wt[(long long)w * 65536 + (long long)col * 256 + f] = f2bf(src[(long long)f * 256 + col]);
}

// ---------------------------------------------------------------------------
// Kernel 2: bf16 WMMA GEMM  Out = X(8192x256) * W(256x256), W transposed.
// vmode==0 : Out in [B][H][N][64]   (Q, K)
// vmode==1 : Out in [B][H][64][N]   (V transposed for attention B-fragments)
// Double-buffered async LDS staging; B-fragments batched 4-wide.
// ---------------------------------------------------------------------------
__global__ void __launch_bounds__(256) k_gemm(
    const unsigned short* __restrict__ X,
    const unsigned short* __restrict__ Wt,
    unsigned short* __restrict__ Out, int vmode)
{
  __shared__ __align__(16) unsigned short Xs[2][128][32];   // 16 KB
  __shared__ __align__(16) unsigned short Ws[2][256][32];   // 32 KB
  const int tid  = threadIdx.x;
  const int lane = tid & 31;
  const int wv   = tid >> 5;
  const int rowbase = blockIdx.x * 128;
  const int hi16 = (lane & 16) ? 1 : 0;
  const unsigned long long Xb = (unsigned long long)(uintptr_t)X;
  const unsigned long long Wb = (unsigned long long)(uintptr_t)Wt;

  v8f acc[16] = {};

  auto stage = [&](int kk, int buf) {
    const int f0 = kk * 32;
    int r = tid >> 1, hf = tid & 1;
    unsigned xl = ldsoff(&Xs[buf][r][hf * 16]);
    unsigned xg = (unsigned)(((size_t)(rowbase + r) * Fdim + f0 + hf * 16) * 2);
    async_b128<0>(xl, xg, Xb);
    async_b128<16>(xl, xg, Xb);
    unsigned wl = ldsoff(&Ws[buf][tid][0]);
    unsigned wg = (unsigned)(((size_t)tid * Fdim + f0) * 2);
    async_b128<0>(wl, wg, Wb);
    async_b128<16>(wl, wg, Wb);
    async_b128<32>(wl, wg, Wb);
    async_b128<48>(wl, wg, Wb);
  };

  stage(0, 0);
  wait_async0();
  __syncthreads();

  for (int kk = 0; kk < Fdim / 32; kk++) {
    const int cur = kk & 1;
    if (kk + 1 < Fdim / 32) stage(kk + 1, cur ^ 1);  // overlap next tile fetch

    // A fragment (16x32): lanes<16 f0..7 & f16..23, lanes>=16 f8..15 & f24..31
    Frag a;
    const char* arow = (const char*)&Xs[cur][wv * 16 + (lane & 15)][0];
    const int aoff = hi16 ? 16 : 0;
    a.q[0] = *(const uint4*)(arow + aoff);
    a.q[1] = *(const uint4*)(arow + 32 + aoff);

    const int boff = hi16 ? 16 : 0;
    auto loadB = [&](Frag& fr, int ct) {
      const char* brow = (const char*)&Ws[cur][ct * 16 + (lane & 15)][0];
      fr.q[0] = *(const uint4*)(brow + boff);
      fr.q[1] = *(const uint4*)(brow + boff + 32);
    };
    #pragma unroll
    for (int g = 0; g < 4; g++) {             // batch 4 B-fragments -> 4 WMMAs
      Frag bq[4];
      #pragma unroll
      for (int j = 0; j < 4; j++) loadB(bq[j], g * 4 + j);
      #pragma unroll
      for (int j = 0; j < 4; j++) acc[g * 4 + j] = wmma_bf16(a, bq[j], acc[g * 4 + j]);
    }
    wait_async0();
    __syncthreads();
  }

  #pragma unroll
  for (int ct = 0; ct < 16; ct++) {
    int col = ct * 16 + (lane & 15);
    int h = col >> 6, c = col & 63;
    #pragma unroll
    for (int v = 0; v < 8; v++) {
      int rg = rowbase + wv * 16 + v + hi16 * 8;
      int b = rg >> 11, n = rg & (Ndim - 1);
      size_t idx = vmode
        ? (((size_t)(b * Hdim + h)) * Cdim + c) * Ndim + n      // [B][H][C][N]
        : (((size_t)(b * Hdim + h)) * Ndim + n) * Cdim + c;     // [B][H][N][C]
      Out[idx] = f2bf(acc[ct][v]);
    }
  }
}

// ---------------------------------------------------------------------------
// Kernel 3: fused masked flash-attention.
// Block = 128 thr = 4 waves = 4 heads; 32 query rows (2 x 16-row tiles).
// S^T = K*Q^T trick: softmax is per-lane; exp(S) lands directly in the
// A-fragment layout for P*V. Double-buffered async K/V/adj staging.
// ---------------------------------------------------------------------------
__global__ void __launch_bounds__(128) k_attn(
    const unsigned short* __restrict__ Q,    // [B][H][N][64]
    const unsigned short* __restrict__ K,    // [B][H][N][64]
    const unsigned short* __restrict__ Vt,   // [B][H][64][N]
    const int*   __restrict__ adj,
    const float* __restrict__ bias,
    float*          __restrict__ outF,
    unsigned short* __restrict__ outBf,
    int mode)                                // 0: relu+bias -> bf16, 1: bias -> f32
{
  __shared__ __align__(16) unsigned short Kt[2][Hdim][32][64]; // 32 KB row-major
  __shared__ __align__(16) unsigned short Vs[2][Hdim][64][32]; // 32 KB [c][m]
  __shared__ int adjT[2][32][32];                              // 8 KB

  const int tid  = threadIdx.x;
  const int lane = tid & 31;
  const int h    = tid >> 5;
  const int b    = blockIdx.x >> 6;          // N/32 = 64 row-blocks per batch
  const int n0   = (blockIdx.x & 63) * 32;
  const int nrow = lane & 15;
  const int hi16 = (lane & 16) ? 1 : 0;
  const int mb   = hi16 * 8;
  const unsigned long long Kbase = (unsigned long long)(uintptr_t)K;
  const unsigned long long Vbase = (unsigned long long)(uintptr_t)Vt;
  const unsigned long long Abase = (unsigned long long)(uintptr_t)adj;

  // Q B-fragments for both 16-row tiles
  Frag qb[2][2];
  #pragma unroll
  for (int u = 0; u < 2; u++) {
    const char* qrow = (const char*)(Q +
        (((size_t)(b * Hdim + h)) * Ndim + n0 + u * 16 + nrow) * Cdim);
    const int off = hi16 ? 16 : 0;
    qb[u][0].q[0] = *(const uint4*)(qrow + off);
    qb[u][0].q[1] = *(const uint4*)(qrow + off + 32);
    qb[u][1].q[0] = *(const uint4*)(qrow + 64 + off);
    qb[u][1].q[1] = *(const uint4*)(qrow + 64 + off + 32);
  }

  v8f   acc[2][4] = {};
  float mrun[2] = {-1e30f, -1e30f}, lrun[2] = {0.f, 0.f};

  const int sr = tid & 31;     // K staging: row within 32-key block
  const int sh = tid >> 5;     // K staging: head

  auto stageA = [&](int ms, int buf) {
    const int m0 = ms * 32;
    { // K rows (row-major, 128 B each)
      unsigned kl = ldsoff(&Kt[buf][sh][sr][0]);
      unsigned kg = (unsigned)((((size_t)(b * Hdim + sh)) * Ndim + m0 + sr) * Cdim * 2);
      async_b128<0>(kl, kg, Kbase);   async_b128<16>(kl, kg, Kbase);
      async_b128<32>(kl, kg, Kbase);  async_b128<48>(kl, kg, Kbase);
      async_b128<64>(kl, kg, Kbase);  async_b128<80>(kl, kg, Kbase);
      async_b128<96>(kl, kg, Kbase);  async_b128<112>(kl, kg, Kbase);
    }
    #pragma unroll
    for (int rr = 0; rr < 2; rr++) { // V^T rows: 256 rows of 64 B, 2 per thread
      int rid = tid * 2 + rr;
      int vh = rid >> 6, vc = rid & 63;
      unsigned vl = ldsoff(&Vs[buf][vh][vc][0]);
      unsigned vg = (unsigned)((((((size_t)(b * Hdim + vh)) * Cdim + vc) * Ndim) + m0) * 2);
      async_b128<0>(vl, vg, Vbase);   async_b128<16>(vl, vg, Vbase);
      async_b128<32>(vl, vg, Vbase);  async_b128<48>(vl, vg, Vbase);
    }
    for (int i = tid; i < 1024; i += 128) {  // adj tile (shared by all heads)
      int rr = i >> 5, mm = i & 31;
      unsigned al = ldsoff(&adjT[buf][rr][mm]);
      unsigned ag = (unsigned)((((size_t)b * Ndim + n0 + rr) * Ndim + m0 + mm) * 4);
      async_b32(al, ag, Abase);
    }
  };

  stageA(0, 0);
  wait_async0();
  __syncthreads();

  for (int ms = 0; ms < Ndim / 32; ms++) {
    const int cur = ms & 1;
    if (ms + 1 < Ndim / 32) stageA(ms + 1, cur ^ 1);   // overlap next block fetch
    if (ms + 2 < Ndim / 32) {                           // L2 prefetch 2 blocks out
      const int m2 = (ms + 2) * 32;
      __builtin_prefetch(K + (((size_t)(b * Hdim + sh)) * Ndim + m2 + sr) * Cdim, 0, 1);
      __builtin_prefetch(Vt + ((((size_t)(b * Hdim + sh)) * Cdim + sr) * Ndim) + m2, 0, 1);
    }

    // ---- batch-load all K A-fragments and V B-fragments ----
    Frag ka[4];  // [m-half][f-half]
    {
      const int aoff = hi16 ? 16 : 0;
      const char* kr0 = (const char*)&Kt[cur][h][nrow][0];
      const char* kr1 = (const char*)&Kt[cur][h][16 + nrow][0];
      ka[0].q[0] = *(const uint4*)(kr0 + aoff);       ka[0].q[1] = *(const uint4*)(kr0 + 32 + aoff);
      ka[1].q[0] = *(const uint4*)(kr0 + 64 + aoff);  ka[1].q[1] = *(const uint4*)(kr0 + 96 + aoff);
      ka[2].q[0] = *(const uint4*)(kr1 + aoff);       ka[2].q[1] = *(const uint4*)(kr1 + 32 + aoff);
      ka[3].q[0] = *(const uint4*)(kr1 + 64 + aoff);  ka[3].q[1] = *(const uint4*)(kr1 + 96 + aoff);
    }
    Frag vb[4];
    #pragma unroll
    for (int t = 0; t < 4; t++) {
      const char* vrow = (const char*)&Vs[cur][h][16 * t + (lane & 15)][0];
      const int boff = hi16 ? 16 : 0;
      vb[t].q[0] = *(const uint4*)(vrow + boff);
      vb[t].q[1] = *(const uint4*)(vrow + boff + 32);
    }

    #pragma unroll
    for (int u = 0; u < 2; u++) {
      // S^T = K_tile * Q^T : lane n & n+16 own query row n0+u*16+n
      v8f s0 = {}, s1 = {};
      s0 = wmma_bf16(ka[0], qb[u][0], s0);
      s0 = wmma_bf16(ka[1], qb[u][1], s0);
      s1 = wmma_bf16(ka[2], qb[u][0], s1);
      s1 = wmma_bf16(ka[3], qb[u][1], s1);

      // masked online softmax
      const int arow = u * 16 + nrow;
      float sv0[8], sv1[8];
      unsigned mbits = 0;
      float tmax = -1e30f;
      #pragma unroll
      for (int i = 0; i < 8; i++) {
        int a0 = adjT[cur][arow][mb + i];
        float xv = s0[i] * 0.125f;
        if (a0 > 0) mbits |= (1u << i); else xv = -1e30f;
        sv0[i] = xv; tmax = fmaxf(tmax, xv);
        int a1 = adjT[cur][arow][16 + mb + i];
        float yv = s1[i] * 0.125f;
        if (a1 > 0) mbits |= (1u << (8 + i)); else yv = -1e30f;
        sv1[i] = yv; tmax = fmaxf(tmax, yv);
      }
      tmax = fmaxf(tmax, __shfl_xor(tmax, 16, 32));
      float mnew  = fmaxf(mrun[u], tmax);
      float alpha = __expf(mrun[u] - mnew);
      float lsum  = 0.f;
      Frag pf;                                 // P already in A-fragment layout
      #pragma unroll
      for (int i = 0; i < 8; i++) {
        float p0 = ((mbits >> i) & 1u)       ? __expf(sv0[i] - mnew) : 0.f;
        float p1 = ((mbits >> (8 + i)) & 1u) ? __expf(sv1[i] - mnew) : 0.f;
        lsum += p0 + p1;
        pf.s[i]     = f2bf(p0);
        pf.s[8 + i] = f2bf(p1);
      }
      lsum += __shfl_xor(lsum, 16, 32);
      lrun[u] = lrun[u] * alpha + lsum;
      mrun[u] = mnew;

      float av[8];
      #pragma unroll
      for (int v = 0; v < 8; v++) av[v] = __shfl(alpha, v + mb, 32);
      #pragma unroll
      for (int t = 0; t < 4; t++)
        #pragma unroll
        for (int v = 0; v < 8; v++) acc[u][t][v] *= av[v];

      #pragma unroll
      for (int t = 0; t < 4; t++)
        acc[u][t] = wmma_bf16(pf, vb[t], acc[u][t]);
    }
    wait_async0();
    __syncthreads();
  }

  // ---- epilogue: normalize, bias, (relu), store ----
  #pragma unroll
  for (int u = 0; u < 2; u++) {
    float lr[8];
    #pragma unroll
    for (int v = 0; v < 8; v++) lr[v] = __shfl(lrun[u], v + mb, 32);
    #pragma unroll
    for (int t = 0; t < 4; t++) {
      int col = 16 * t + (lane & 15);
      float bv = bias[h * 64 + col];
      #pragma unroll
      for (int v = 0; v < 8; v++) {
        int row = n0 + u * 16 + v + mb;
        float val = acc[u][t][v] / (lr[v] + 1e-16f) + bv;
        size_t idx = ((size_t)(b * Ndim) + row) * HCdim + h * 64 + col;
        if (mode == 0) outBf[idx] = f2bf(fmaxf(val, 0.f));
        else           outF[idx]  = val;
      }
    }
  }
}

// ---------------------------------------------------------------------------
extern "C" void kernel_launch(void* const* d_in, const int* in_sizes, int n_in,
                              void* d_out, int out_size, void* d_ws, size_t ws_size,
                              hipStream_t stream)
{
  const float* x   = (const float*)d_in[0];
  const int*   adj = (const int*)  d_in[1];
  const float* Wq1 = (const float*)d_in[2];
  const float* Wk1 = (const float*)d_in[3];
  const float* Wv1 = (const float*)d_in[4];
  const float* b1  = (const float*)d_in[5];
  const float* Wq2 = (const float*)d_in[6];
  const float* Wk2 = (const float*)d_in[7];
  const float* Wv2 = (const float*)d_in[8];
  const float* b2  = (const float*)d_in[9];
  float* out = (float*)d_out;

  unsigned short* ws = (unsigned short*)d_ws;
  const size_t NX = (size_t)Bdim * Ndim * Fdim;         // 2,097,152 (== B*H*N*C)
  unsigned short* xbf = ws;
  unsigned short* wt  = xbf + NX;
  unsigned short* Qb  = wt + 6 * 65536;
  unsigned short* Kb  = Qb + NX;
  unsigned short* Vb  = Kb + NX;
  unsigned short* hbf = Vb + NX;

  const int convThreads = (int)(NX + 6 * 65536);
  k_convert<<<convThreads / 256, 256, 0, stream>>>(x, Wq1, Wk1, Wv1, Wq2, Wk2, Wv2, xbf, wt);

  // layer 1
  k_gemm<<<64, 256, 0, stream>>>(xbf, wt + 0 * 65536, Qb, 0);
  k_gemm<<<64, 256, 0, stream>>>(xbf, wt + 1 * 65536, Kb, 0);
  k_gemm<<<64, 256, 0, stream>>>(xbf, wt + 2 * 65536, Vb, 1);
  k_attn<<<Bdim * (Ndim / 32), 128, 0, stream>>>(Qb, Kb, Vb, adj, b1, nullptr, hbf, 0);

  // layer 2
  k_gemm<<<64, 256, 0, stream>>>(hbf, wt + 3 * 65536, Qb, 0);
  k_gemm<<<64, 256, 0, stream>>>(hbf, wt + 4 * 65536, Kb, 0);
  k_gemm<<<64, 256, 0, stream>>>(hbf, wt + 5 * 65536, Vb, 1);
  k_attn<<<Bdim * (Ndim / 32), 128, 0, stream>>>(Qb, Kb, Vb, adj, b2, out, nullptr, 1);
}
